// PennesHPM_84018150245203
// MI455X (gfx1250) — matrix operand order repaired and checked by reading, not compiled, not run on hardware
//
#include <hip/hip_runtime.h>
#include <math.h>

typedef _Float16 half_t;
typedef _Float16 v16h __attribute__((ext_vector_type(16)));
typedef _Float16 v8h  __attribute__((ext_vector_type(8)));
typedef __fp16   f16x2 __attribute__((ext_vector_type(2)));   // cvt_pkrtz return type
typedef float    v8f  __attribute__((ext_vector_type(8)));
typedef unsigned int uint4v __attribute__((ext_vector_type(4)));

#define HID 64
#define WTAB 480
#define TWOPI 6.28318530717958647692f

__device__ __forceinline__ float fast_tanh(float x) {
#if __has_builtin(__builtin_amdgcn_tanhf)
  return __builtin_amdgcn_tanhf(x);
#else
  return tanhf(x);
#endif
}

__device__ __forceinline__ v8f wmma_f16(v16h a, v16h b, v8f c) {
  return __builtin_amdgcn_wmma_f32_16x16x32_f16(false, a, false, b, (short)0, c, false, false);
}

// v_cvt_pk_rtz_f16_f32: pack two f32 into one dword of two f16
__device__ __forceinline__ unsigned int pk2(float a, float b) {
  f16x2 p = __builtin_amdgcn_cvt_pkrtz(a, b);
  return __builtin_bit_cast(unsigned int, p);
}

// ---------------------------------------------------------------------------
// Transposed formulation: H^T = W^T x X^T. Weights live on the A side
// (register-resident fragments, loaded once per persistent wave); activations
// on the B side, where the D layout (lane = point column, rows = hidden) maps
// to [point][hidden]-contiguous LDS: 4x v_cvt_pk + 1x ds_store_b128 per acc,
// 2x ds_load_b128 per B fragment. Both 16-point tiles advance in lockstep per
// layer so 8/16/4 independent WMMAs are in flight between hazards, with one
// s_wait_dscnt per layer transition. Layer 3 replicates W3 across all A rows,
// so heat[point] is read directly from the accumulator.
// ---------------------------------------------------------------------------
__global__ void __launch_bounds__(256) pennes_main(
    const float* __restrict__ der,
    const float* __restrict__ a1, const float* __restrict__ a2,
    const float* __restrict__ a3, const float* __restrict__ a4,
    const float* __restrict__ a5, const float* __restrict__ a6,
    const float* __restrict__ a7, const float* __restrict__ a9,
    const float* __restrict__ W1, const float* __restrict__ b1,
    const float* __restrict__ W2, const float* __restrict__ b2,
    const float* __restrict__ W3, const float* __restrict__ b3,
    float* __restrict__ out, int ntiles)
{
  // per-wave LDS: H^T tile, 32 points x 64 hidden f16, row stride padded to
  // 72 halves (144 B -> 36 dwords, gcd(36,64)=4: clean bank spread)
  __shared__ __align__(16) half_t sH[8][32 * 72];

  const int lane  = threadIdx.x & 31;
  const int wid   = threadIdx.x >> 5;
  const int lmod  = lane & 15;
  const int lhalf = lane >> 4;

  // element e of an A fragment holds K = (e<8 ? lhalf*8+e : 16+lhalf*8+e-8)
  int kmapv[16];
#pragma unroll
  for (int e = 0; e < 16; ++e)
    kmapv[e] = (e < 8) ? (lhalf * 8 + e) : (16 + lhalf * 8 + (e - 8));

  // ---- register-resident weight A-fragments (W^T), loaded once ----
  v16h w1f[4];        // W1^T: rows = hidden tile, K = input dim padded 3->32
  v16h w2f[4][2];     // W2^T: [out-tile][K-tile]
  v16h w3f[2];        // W3^T replicated across all 16 rows
  for (int jt = 0; jt < 4; ++jt) {
    const int h = jt * 16 + lmod;
    for (int e = 0; e < 16; ++e) {
      const int K = kmapv[e];
      w1f[jt][e] = (half_t)((K < 3) ? W1[K * HID + h] : 0.0f);
    }
  }
  for (int jt = 0; jt < 4; ++jt) {
    const int h = jt * 16 + lmod;
    for (int kt = 0; kt < 2; ++kt)
      for (int e = 0; e < 16; ++e) {
        const int K = kt * 32 + kmapv[e];
        w2f[jt][kt][e] = (half_t)W2[K * HID + h];
      }
  }
  for (int kt = 0; kt < 2; ++kt)
    for (int e = 0; e < 16; ++e)
      w3f[kt][e] = (half_t)W3[kt * 32 + kmapv[e]];   // same for every row

  // ---- bias accumulators (D layout: VGPR r -> hidden row lhalf*8+r) ----
  v8f b1acc[4], b2acc[4], b3acc;
  for (int jt = 0; jt < 4; ++jt)
    for (int r = 0; r < 8; ++r) {
      b1acc[jt][r] = b1[jt * 16 + lhalf * 8 + r];
      b2acc[jt][r] = b2[jt * 16 + lhalf * 8 + r];
    }
  for (int r = 0; r < 8; ++r) b3acc[r] = b3[0];

  const int gw      = blockIdx.x * (blockDim.x >> 5) + wid;
  const int gstride = gridDim.x * (blockDim.x >> 5);
  half_t* myH = sH[wid];

  for (int tile = gw; tile < ntiles; tile += gstride) {
    const int p = tile * 32 + lane;
    const float x   = der[p * 9 + 0];
    const float y   = der[p * 9 + 1];
    const float t   = der[p * 9 + 2];
    const float xi  = der[p * 9 + 3];
    const float yi  = der[p * 9 + 4];
    const float u   = der[p * 9 + 5];
    const float uxx = der[p * 9 + 6];
    const float uyy = der[p * 9 + 7];
    __builtin_prefetch(&der[(size_t)(p + gstride * 32) * 9], 0, 1);

    // layer-1 B = X^T (K=3 padded): lanes 0-15 carry K=0..2, upper half zero
    const float xo = __shfl_xor(x, 16, 32);
    const float yo = __shfl_xor(y, 16, 32);
    const float to = __shfl_xor(t, 16, 32);

    v16h bx[2];
    bx[0] = (v16h){};
    bx[1] = (v16h){};
    bx[0][0] = (half_t)(lhalf ? 0.0f : x);
    bx[0][1] = (half_t)(lhalf ? 0.0f : y);
    bx[0][2] = (half_t)(lhalf ? 0.0f : t);
    bx[1][0] = (half_t)(lhalf ? 0.0f : xo);
    bx[1][1] = (half_t)(lhalf ? 0.0f : yo);
    bx[1][2] = (half_t)(lhalf ? 0.0f : to);

    // ================= layer 1: 8 independent WMMAs =================
    v8f h[2][4];
#pragma unroll
    for (int pt = 0; pt < 2; ++pt)
#pragma unroll
      for (int jt = 0; jt < 4; ++jt)
        h[pt][jt] = wmma_f16(w1f[jt], bx[pt], b1acc[jt]);
#pragma unroll
    for (int pt = 0; pt < 2; ++pt)
#pragma unroll
      for (int jt = 0; jt < 4; ++jt)
#pragma unroll
        for (int r = 0; r < 8; ++r) h[pt][jt][r] = fast_tanh(h[pt][jt][r]);

    // pack + store H1^T: [point][hidden], 8 contiguous halves per acc
#pragma unroll
    for (int pt = 0; pt < 2; ++pt)
#pragma unroll
      for (int jt = 0; jt < 4; ++jt) {
        uint4v sv = {pk2(h[pt][jt][0], h[pt][jt][1]), pk2(h[pt][jt][2], h[pt][jt][3]),
                     pk2(h[pt][jt][4], h[pt][jt][5]), pk2(h[pt][jt][6], h[pt][jt][7])};
        *(uint4v*)&myH[(pt * 16 + lmod) * 72 + jt * 16 + lhalf * 8] = sv;
      }
    asm volatile("s_wait_dscnt 0" ::: "memory");

    // load H1^T as B fragments (K in table order: lhalf*8+e, 16+lhalf*8+e-8)
    v16h B2[2][2];
#pragma unroll
    for (int pt = 0; pt < 2; ++pt)
#pragma unroll
      for (int kk = 0; kk < 2; ++kk) {
        v8h lo = *(const v8h*)&myH[(pt * 16 + lmod) * 72 + kk * 32 + lhalf * 8];
        v8h hi = *(const v8h*)&myH[(pt * 16 + lmod) * 72 + kk * 32 + 16 + lhalf * 8];
        B2[pt][kk] = __builtin_shufflevector(lo, hi, 0, 1, 2, 3, 4, 5, 6, 7,
                                                     8, 9, 10, 11, 12, 13, 14, 15);
      }

    // ================= layer 2: 16 WMMAs (8 chains of 2) =================
    v8f g[2][4];
#pragma unroll
    for (int pt = 0; pt < 2; ++pt)
#pragma unroll
      for (int jt = 0; jt < 4; ++jt)
        g[pt][jt] = wmma_f16(w2f[jt][0], B2[pt][0], b2acc[jt]);
#pragma unroll
    for (int pt = 0; pt < 2; ++pt)
#pragma unroll
      for (int jt = 0; jt < 4; ++jt)
        g[pt][jt] = wmma_f16(w2f[jt][1], B2[pt][1], g[pt][jt]);
#pragma unroll
    for (int pt = 0; pt < 2; ++pt)
#pragma unroll
      for (int jt = 0; jt < 4; ++jt)
#pragma unroll
        for (int r = 0; r < 8; ++r) g[pt][jt][r] = fast_tanh(g[pt][jt][r]);

#pragma unroll
    for (int pt = 0; pt < 2; ++pt)
#pragma unroll
      for (int jt = 0; jt < 4; ++jt) {
        uint4v sv = {pk2(g[pt][jt][0], g[pt][jt][1]), pk2(g[pt][jt][2], g[pt][jt][3]),
                     pk2(g[pt][jt][4], g[pt][jt][5]), pk2(g[pt][jt][6], g[pt][jt][7])};
        *(uint4v*)&myH[(pt * 16 + lmod) * 72 + jt * 16 + lhalf * 8] = sv;
      }
    asm volatile("s_wait_dscnt 0" ::: "memory");

    v16h B3[2][2];
#pragma unroll
    for (int pt = 0; pt < 2; ++pt)
#pragma unroll
      for (int kk = 0; kk < 2; ++kk) {
        v8h lo = *(const v8h*)&myH[(pt * 16 + lmod) * 72 + kk * 32 + lhalf * 8];
        v8h hi = *(const v8h*)&myH[(pt * 16 + lmod) * 72 + kk * 32 + 16 + lhalf * 8];
        B3[pt][kk] = __builtin_shufflevector(lo, hi, 0, 1, 2, 3, 4, 5, 6, 7,
                                                     8, 9, 10, 11, 12, 13, 14, 15);
      }

    // ===== layer 3: 4 WMMAs; W3 replicated -> every D row = heat[point] ====
    v8f c30 = wmma_f16(w3f[0], B3[0][0], b3acc);
    v8f c31 = wmma_f16(w3f[0], B3[1][0], b3acc);
    c30 = wmma_f16(w3f[1], B3[0][1], c30);
    c31 = wmma_f16(w3f[1], B3[1][1], c31);

    // lane's own point: lanes 0-15 -> tile 0, lanes 16-31 -> tile 1
    const float heat = lhalf ? c31[0] : c30[0];

    // ---- elementwise Pennes terms (tables L2-resident) ----
    const int gidx = (int)xi * WTAB + (int)yi;
    const float g1 = a1[gidx], g2 = a2[gidx], g3 = a3[gidx], g4 = a4[gidx];
    const float g5 = a5[gidx], g6 = a6[gidx], g7 = a7[gidx], g9 = a9[gidx];

    const float conv  = 0.12f * fmaxf(g5, 0.0f) * (uxx + uyy);
    const float perf  = fmaxf(g1, 0.0f) * (37.0f - u);
    const float metab = 0.003f * fmaxf(g4, 0.0f) * __expf((u - 37.0f) / 10.0f);
    const float resp  = g2 * __sinf(TWOPI * 0.1f * t + g3);
    const float heart = g6 * __sinf(TWOPI * 0.25f * t + g7);
    const float cool  = fmaxf(g9, 0.0f) * (21.0f - u);

    out[p] = conv + perf + resp + heat + metab + heart + cool;
  }
}

// ---------------------------------------------------------------------------
// Scalar tail kernel (N % 32 points; N=2M divides evenly so normally unused)
// ---------------------------------------------------------------------------
__global__ void pennes_tail(
    const float* __restrict__ der,
    const float* __restrict__ a1, const float* __restrict__ a2,
    const float* __restrict__ a3, const float* __restrict__ a4,
    const float* __restrict__ a5, const float* __restrict__ a6,
    const float* __restrict__ a7, const float* __restrict__ a9,
    const float* __restrict__ W1, const float* __restrict__ b1,
    const float* __restrict__ W2, const float* __restrict__ b2,
    const float* __restrict__ W3, const float* __restrict__ b3,
    float* __restrict__ out, int start, int n)
{
  const int p = start + blockIdx.x * blockDim.x + threadIdx.x;
  if (p >= n) return;
  const float x   = der[p * 9 + 0];
  const float y   = der[p * 9 + 1];
  const float t   = der[p * 9 + 2];
  const float xi  = der[p * 9 + 3];
  const float yi  = der[p * 9 + 4];
  const float u   = der[p * 9 + 5];
  const float uxx = der[p * 9 + 6];
  const float uyy = der[p * 9 + 7];

  float h1[HID], h2[HID];
  for (int k = 0; k < HID; ++k)
    h1[k] = fast_tanh(b1[k] + x * W1[k] + y * W1[HID + k] + t * W1[2 * HID + k]);
  for (int k = 0; k < HID; ++k) {
    float s = b2[k];
    for (int j = 0; j < HID; ++j) s += h1[j] * W2[j * HID + k];
    h2[k] = fast_tanh(s);
  }
  float heat = b3[0];
  for (int k = 0; k < HID; ++k) heat += h2[k] * W3[k];

  const int gidx = (int)xi * WTAB + (int)yi;
  const float conv  = 0.12f * fmaxf(a5[gidx], 0.0f) * (uxx + uyy);
  const float perf  = fmaxf(a1[gidx], 0.0f) * (37.0f - u);
  const float metab = 0.003f * fmaxf(a4[gidx], 0.0f) * __expf((u - 37.0f) / 10.0f);
  const float resp  = a2[gidx] * __sinf(TWOPI * 0.1f * t + a3[gidx]);
  const float heart = a6[gidx] * __sinf(TWOPI * 0.25f * t + a7[gidx]);
  const float cool  = fmaxf(a9[gidx], 0.0f) * (21.0f - u);
  out[p] = conv + perf + resp + heat + metab + heart + cool;
}

extern "C" void kernel_launch(void* const* d_in, const int* in_sizes, int n_in,
                              void* d_out, int out_size, void* d_ws, size_t ws_size,
                              hipStream_t stream) {
  (void)n_in; (void)d_ws; (void)ws_size; (void)out_size;
  const int N = in_sizes[0] / 9;
  const float* der = (const float*)d_in[0];
  const float* a1 = (const float*)d_in[1];
  const float* a2 = (const float*)d_in[2];
  const float* a3 = (const float*)d_in[3];
  const float* a4 = (const float*)d_in[4];
  const float* a5 = (const float*)d_in[5];
  const float* a6 = (const float*)d_in[6];
  const float* a7 = (const float*)d_in[7];
  // d_in[8] = a_8 (unused by the reference math)
  const float* a9 = (const float*)d_in[9];
  const float* W1 = (const float*)d_in[10];
  const float* b1 = (const float*)d_in[11];
  const float* W2 = (const float*)d_in[12];
  const float* b2 = (const float*)d_in[13];
  const float* W3 = (const float*)d_in[14];
  const float* b3 = (const float*)d_in[15];
  float* out = (float*)d_out;

  const int ntiles = N / 32;
  if (ntiles > 0) {
    int blocks = 512;                       // persistent waves, grid-stride
    if (blocks * 8 > ntiles) blocks = (ntiles + 7) / 8;
    pennes_main<<<blocks, 256, 0, stream>>>(der, a1, a2, a3, a4, a5, a6, a7, a9,
                                            W1, b1, W2, b2, W3, b3, out, ntiles);
  }
  const int tail = N - ntiles * 32;
  if (tail > 0) {
    pennes_tail<<<(tail + 63) / 64, 64, 0, stream>>>(der, a1, a2, a3, a4, a5, a6,
                                                     a7, a9, W1, b1, W2, b2, W3,
                                                     b3, out, ntiles * 32, N);
  }
}
